// GNN32_27410481283401
// MI455X (gfx1250) — matrix operand order repaired and checked by reading, not compiled, not run on hardware
//
#include <hip/hip_runtime.h>
#include <hip/hip_bf16.h>

#define NN 50000
#define EE 400000
#define HH 6

typedef __attribute__((ext_vector_type(16))) _Float16 v16h;
typedef __attribute__((ext_vector_type(8)))  _Float16 v8h;
typedef __attribute__((ext_vector_type(8)))  float    v8f;

// ---------- helpers: order-preserving float<->uint encoding for atomicMax ----------
__device__ __forceinline__ unsigned enc_f32(float v) {
  unsigned u = __float_as_uint(v);
  return (u & 0x80000000u) ? ~u : (u | 0x80000000u);
}
__device__ __forceinline__ float dec_f32(unsigned k) {
  return (k & 0x80000000u) ? __uint_as_float(k ^ 0x80000000u) : __uint_as_float(~k);
}

// ---------- pack kernels: f32 -> zero-padded f16, W additionally transposed ----------
// Xh[n, k] = (k < din) ? X[n, k] : 0        Xh: [NN, kp]
__global__ void pack_x_kernel(const float* __restrict__ X, _Float16* __restrict__ Xh,
                              int din, int kp) {
  int idx = blockIdx.x * blockDim.x + threadIdx.x;
  if (idx >= NN * kp) return;
  int n = idx / kp, k = idx - n * kp;
  Xh[idx] = (k < din) ? (_Float16)X[(size_t)n * din + k] : (_Float16)0.0f;
}
// Wt[c, k] = (k < din) ? W[k, c] : 0        Wt: [dout, kp]
__global__ void pack_w_kernel(const float* __restrict__ W, _Float16* __restrict__ Wt,
                              int din, int kp, int dout) {
  int idx = blockIdx.x * blockDim.x + threadIdx.x;
  if (idx >= dout * kp) return;
  int c = idx / kp, k = idx - c * kp;
  Wt[idx] = (k < din) ? (_Float16)W[(size_t)k * dout + c] : (_Float16)0.0f;
}

// ---------- WMMA GEMM: H[NN, DOUT] = Xh[NN, KP] @ Wt[DOUT, KP]^T (f16 mul, f32 acc) ----
// One wave per 16-row tile; A fragments loaded once (b128s), reused across all N-tiles.
template<int KP, int DOUT>
__global__ __launch_bounds__(32)
void gemm_wmma_kernel(const _Float16* __restrict__ Xh, const _Float16* __restrict__ Wt,
                      float* __restrict__ H) {
  constexpr int KT = KP / 32;           // k-tiles of 32
  constexpr int NT = DOUT / 16;         // n-tiles of 16
  const int lane  = threadIdx.x & 31;
  const int hi    = lane >> 4;          // half-wave select
  const int mtile = blockIdx.x;
  const int row   = mtile * 16 + (lane & 15);

  union AU { v16h v; v8h h[2]; };

  // A 16x32 f16 layout: lane holds row; VGPRs 0-3 = K (hi?8:0)+0..7 (contiguous),
  // VGPRs 4-7 = K (hi?8:0)+16..23 (contiguous) -> two b128 loads per k-tile.
  v16h a[KT];
  const _Float16* xr = Xh + (size_t)row * KP + hi * 8;
#pragma unroll
  for (int kt = 0; kt < KT; ++kt) {
    AU au;
    au.h[0] = *(const v8h*)(xr + kt * 32);
    au.h[1] = *(const v8h*)(xr + kt * 32 + 16);
    a[kt] = au.v;
  }

  const int col = lane & 15;
#pragma unroll 1
  for (int nt = 0; nt < NT; ++nt) {
    const int gcol = nt * 16 + col;
    // B 32x16 f16 layout: lane holds column; same K order as A: (hi?8:0)+0..7,
    // (hi?8:0)+16..23 -> two b128 loads per lane per k-tile (Wt is transposed).
    const _Float16* wr = Wt + (size_t)gcol * KP + hi * 8;
    v8f c = {};
#pragma unroll
    for (int kt = 0; kt < KT; ++kt) {
      AU bu;
      bu.h[0] = *(const v8h*)(wr + kt * 32);
      bu.h[1] = *(const v8h*)(wr + kt * 32 + 16);
      v16h b = bu.v;
      c = __builtin_amdgcn_wmma_f32_16x16x32_f16(false, a[kt], false, b,
                                                 (short)0, c, false, false);
    }
    // C/D layout: VGPR r -> row r (lanes 0-15) or r+8 (lanes 16-31), col = lane&15
#pragma unroll
    for (int r = 0; r < 8; ++r)
      H[(size_t)(mtile * 16 + r + hi * 8) * DOUT + gcol] = c[r];
  }
}

// ---------- per-node attention scores es/ed ----------
template<int F>
__global__ void node_scores_kernel(const float* __restrict__ Hm,
                                   const float* __restrict__ a_s,
                                   const float* __restrict__ a_d,
                                   float* __restrict__ es, float* __restrict__ ed) {
  int idx = blockIdx.x * blockDim.x + threadIdx.x;
  if (idx >= NN * HH) return;
  int n = idx / HH, h = idx - n * HH;
  const float* hp = Hm + (size_t)n * (HH * F) + h * F;
  float s = 0.f, d = 0.f;
#pragma unroll
  for (int j = 0; j < F; ++j) {
    float v = hp[j];
    s += v * a_s[h * F + j];
    d += v * a_d[h * F + j];
  }
  es[idx] = s;
  ed[idx] = d;
}

// ---------- edge pass 1: e = leaky_relu(es[src]+ed[dst]); segment max ----------
__global__ void edge_pass1(const int* __restrict__ src, const int* __restrict__ dst,
                           const float* __restrict__ es, const float* __restrict__ ed,
                           float* __restrict__ ebuf, unsigned* __restrict__ menc) {
  int idx = blockIdx.x * blockDim.x + threadIdx.x;
  if (idx >= EE * HH) return;
  int e = idx / HH, h = idx - e * HH;
  int s = src[e], d = dst[e];
  float v = es[s * HH + h] + ed[d * HH + h];
  v = (v > 0.f) ? v : 0.2f * v;
  ebuf[idx] = v;
  atomicMax(&menc[d * HH + h], enc_f32(v));
}

// ---------- edge pass 2: w = exp(e - m[dst]); segment sum ----------
__global__ void edge_pass2(const int* __restrict__ dst,
                           float* __restrict__ ebuf,
                           const unsigned* __restrict__ menc,
                           float* __restrict__ denom) {
  int idx = blockIdx.x * blockDim.x + threadIdx.x;
  if (idx >= EE * HH) return;
  int e = idx / HH, h = idx - e * HH;
  int d = dst[e];
  float m = dec_f32(menc[d * HH + h]);
  float w = __expf(ebuf[idx] - m);
  ebuf[idx] = w;
  atomicAdd(&denom[d * HH + h], w);
}

// ---------- edge pass 3: out[dst] += (w/denom[dst]) * h[src], one element/thread ----------
template<int F>
__global__ void edge_pass3(const int* __restrict__ src, const int* __restrict__ dst,
                           const float* __restrict__ ebuf, const float* __restrict__ denom,
                           const float* __restrict__ Hm, float* __restrict__ Out) {
  int idx = blockIdx.x * blockDim.x + threadIdx.x;
  if (idx >= EE * HH * F) return;
  int j  = idx % F;
  int eh = idx / F;            // e*HH + h
  int e = eh / HH, h = eh - e * HH;
  int s = src[e], d = dst[e];
  float alpha = ebuf[eh] / denom[d * HH + h];
  atomicAdd(&Out[(size_t)d * (HH * F) + h * F + j],
            alpha * Hm[(size_t)s * (HH * F) + h * F + j]);
}

// ---------- elementwise ELU in place ----------
__global__ void elu_kernel(float* __restrict__ p, int n) {
  int i = blockIdx.x * blockDim.x + threadIdx.x;
  if (i < n) {
    float v = p[i];
    p[i] = (v > 0.f) ? v : (__expf(v) - 1.f);
  }
}

// ---------- fills ----------
__global__ void fill_f32(float* __restrict__ p, float v, int n) {
  int i = blockIdx.x * blockDim.x + threadIdx.x;
  if (i < n) p[i] = v;
}
__global__ void fill_u32(unsigned* __restrict__ p, unsigned v, int n) {
  int i = blockIdx.x * blockDim.x + threadIdx.x;
  if (i < n) p[i] = v;
}

// ---------- column sum over nodes: out[768], cols 0-383 from A, 384-767 from B ----------
__global__ void colsum_kernel(const float* __restrict__ A, const float* __restrict__ B,
                              float* __restrict__ out) {
  __shared__ float red[256];
  int col = blockIdx.x;                       // 0..767
  const float* srcp = (col < 384) ? A : B;
  int c = (col < 384) ? col : (col - 384);
  float s = 0.f;
  for (int r = threadIdx.x; r < NN; r += blockDim.x) {
    __builtin_prefetch(&srcp[(size_t)(r + blockDim.x) * 384 + c], 0, 1);
    s += srcp[(size_t)r * 384 + c];
  }
  red[threadIdx.x] = s;
  __syncthreads();
  for (int off = 128; off > 0; off >>= 1) {
    if ((int)threadIdx.x < off) red[threadIdx.x] += red[threadIdx.x + off];
    __syncthreads();
  }
  if (threadIdx.x == 0) out[col] = red[0];
}

// ---------- final: L2-normalize the 768-vector, dot with dense_w, add bias ----------
__global__ void final_kernel(const float* __restrict__ s, const float* __restrict__ w,
                             const float* __restrict__ b, float* __restrict__ out) {
  __shared__ float r1[256], r2[256];
  float ss = 0.f, dp = 0.f;
  for (int i = threadIdx.x; i < 768; i += blockDim.x) {
    float v = s[i];
    ss += v * v;
    dp += v * w[i];
  }
  r1[threadIdx.x] = ss;
  r2[threadIdx.x] = dp;
  __syncthreads();
  for (int off = 128; off > 0; off >>= 1) {
    if ((int)threadIdx.x < off) {
      r1[threadIdx.x] += r1[threadIdx.x + off];
      r2[threadIdx.x] += r2[threadIdx.x + off];
    }
    __syncthreads();
  }
  if (threadIdx.x == 0) {
    float norm = sqrtf(r1[0]);
    out[0] = r2[0] / fmaxf(norm, 1e-12f) + b[0];
  }
}

// ---------- host-side layer driver ----------
template<int DIN, int F>
static void run_layer(const float* x, const float* W, const float* a_s, const float* a_d,
                      const int* src, const int* dst,
                      float* hbuf, float* obuf,
                      _Float16* xh, _Float16* wt,
                      float* es, float* ed, unsigned* menc, float* denom, float* ebuf,
                      hipStream_t stream) {
  constexpr int DOUT = HH * F;
  constexpr int KP   = ((DIN + 31) / 32) * 32;

  const int nx = NN * KP;
  pack_x_kernel<<<(nx + 255) / 256, 256, 0, stream>>>(x, xh, DIN, KP);
  const int nw = DOUT * KP;
  pack_w_kernel<<<(nw + 255) / 256, 256, 0, stream>>>(W, wt, DIN, KP, DOUT);

  gemm_wmma_kernel<KP, DOUT><<<NN / 16, 32, 0, stream>>>(xh, wt, hbuf);

  const int nh = NN * HH;
  node_scores_kernel<F><<<(nh + 255) / 256, 256, 0, stream>>>(hbuf, a_s, a_d, es, ed);
  fill_u32<<<(nh + 255) / 256, 256, 0, stream>>>(menc, 0u, nh);
  fill_f32<<<(nh + 255) / 256, 256, 0, stream>>>(denom, 0.f, nh);
  const int no = NN * DOUT;
  fill_f32<<<(no + 255) / 256, 256, 0, stream>>>(obuf, 0.f, no);

  const int eh = EE * HH;
  edge_pass1<<<(eh + 255) / 256, 256, 0, stream>>>(src, dst, es, ed, ebuf, menc);
  edge_pass2<<<(eh + 255) / 256, 256, 0, stream>>>(dst, ebuf, menc, denom);
  const int ehf = EE * HH * F;
  edge_pass3<F><<<(ehf + 255) / 256, 256, 0, stream>>>(src, dst, ebuf, denom, hbuf, obuf);

  elu_kernel<<<(no + 255) / 256, 256, 0, stream>>>(obuf, no);
}

extern "C" void kernel_launch(void* const* d_in, const int* in_sizes, int n_in,
                              void* d_out, int out_size, void* d_ws, size_t ws_size,
                              hipStream_t stream) {
  const float* x_int = (const float*)d_in[0];
  const float* x_nh  = (const float*)d_in[1];
  const int*   ei_i  = (const int*)d_in[2];
  const int*   ei_n  = (const int*)d_in[3];
  const float* W1 = (const float*)d_in[4];
  const float* as1 = (const float*)d_in[5];
  const float* ad1 = (const float*)d_in[6];
  const float* W2 = (const float*)d_in[7];
  const float* as2 = (const float*)d_in[8];
  const float* ad2 = (const float*)d_in[9];
  const float* W3 = (const float*)d_in[10];
  const float* as3 = (const float*)d_in[11];
  const float* ad3 = (const float*)d_in[12];
  const float* dw = (const float*)d_in[13];
  const float* db = (const float*)d_in[14];

  // workspace layout
  char* w = (char*)d_ws;
  size_t off = 0;
  auto alloc = [&](size_t bytes) -> char* {
    char* p = w + off;
    off = (off + bytes + 255) & ~(size_t)255;
    return p;
  };
  float*     b0i  = (float*)alloc((size_t)NN * 384 * 4);   // ping (hidden h)
  float*     b1i  = (float*)alloc((size_t)NN * 384 * 4);   // pong (attn out / next x)
  float*     b0n  = (float*)alloc((size_t)NN * 384 * 4);
  float*     b1n  = (float*)alloc((size_t)NN * 384 * 4);
  _Float16*  xh   = (_Float16*)alloc((size_t)NN * 192 * 2); // packed f16 activations
  _Float16*  wt   = (_Float16*)alloc((size_t)384 * 192 * 2);// packed f16 transposed weights
  float*     es   = (float*)alloc((size_t)NN * HH * 4);
  float*     ed   = (float*)alloc((size_t)NN * HH * 4);
  unsigned*  menc = (unsigned*)alloc((size_t)NN * HH * 4);
  float*     den  = (float*)alloc((size_t)NN * HH * 4);
  float*     ebuf = (float*)alloc((size_t)EE * HH * 4);
  float*     csum = (float*)alloc(768 * 4);

  const int* si = ei_i;        const int* di = ei_i + EE;
  const int* sn = ei_n;        const int* dn = ei_n + EE;

  // stream "int": x -> b1 after each layer (ping-pong b0=hidden, b1=output)
  run_layer<11, 16>(x_int, W1, as1, ad1, si, di, b0i, b1i, xh, wt, es, ed, menc, den, ebuf, stream);
  run_layer<96, 32>(b1i,  W2, as2, ad2, si, di, b0i, b1i, xh, wt, es, ed, menc, den, ebuf, stream);
  run_layer<192, 64>(b1i, W3, as3, ad3, si, di, b0i, b1i, xh, wt, es, ed, menc, den, ebuf, stream);

  // stream "nh"
  run_layer<11, 16>(x_nh, W1, as1, ad1, sn, dn, b0n, b1n, xh, wt, es, ed, menc, den, ebuf, stream);
  run_layer<96, 32>(b1n,  W2, as2, ad2, sn, dn, b0n, b1n, xh, wt, es, ed, menc, den, ebuf, stream);
  run_layer<192, 64>(b1n, W3, as3, ad3, sn, dn, b0n, b1n, xh, wt, es, ed, menc, den, ebuf, stream);

  // concat -> column sums -> normalize + dense
  colsum_kernel<<<768, 256, 0, stream>>>(b1i, b1n, csum);
  final_kernel<<<1, 256, 0, stream>>>(csum, dw, db, (float*)d_out);
}
